// MultiHeadSparseAttention_57578331570456
// MI455X (gfx1250) — matrix operand order repaired and checked by reading, not compiled, not run on hardware
//
#include <hip/hip_runtime.h>
#include <hip/hip_bf16.h>

// MI455X / gfx1250, wave32. Matrix ops via v_wmma_f32_16x16x32_f16;
// dense attn rows drained LDS->HBM via global_store_async_from_lds_b128.

typedef __attribute__((ext_vector_type(16))) _Float16 v16h;
typedef __attribute__((ext_vector_type(8)))  float    v8f;
typedef __attribute__((ext_vector_type(4)))  int      v4i;

#define DMODEL 1024
#define SEQ    1024
#define BATCH  4
#define NHEADS 16
#define DEPTH  64
#define GWIDTH 32

#if defined(__has_builtin)
#if __has_builtin(__builtin_amdgcn_global_store_async_from_lds_b128) && \
    __has_builtin(__builtin_amdgcn_s_wait_asynccnt)
#define USE_ASYNC_ATTN 1
#endif
#endif
#ifndef USE_ASYNC_ATTN
#define USE_ASYNC_ATTN 0
#endif

// ---------------------------------------------------------------------------
// GEMM: C[M x 1024] = A[M x 1024] * W[1024 x 1024] + bias, fp32 in/out,
// f16 WMMA with fp32 accumulation. Block = 256 threads (8 waves),
// tile 128(M) x 64(N), K-blocked by 32. Each wave owns a 16-row strip and
// 4 N-tiles: 4 WMMAs per barrier interval with a shared A fragment.
// blockIdx.z selects among three (W,bias,C) triples (fused Q/K/V launch).
// LDS pitch = 40 halves keeps every fragment chunk 16B-aligned.
// ---------------------------------------------------------------------------
__global__ __launch_bounds__(256) void gemm_f16_wmma(
    const float* __restrict__ A,
    const float* __restrict__ W0, const float* __restrict__ b0, float* __restrict__ C0,
    const float* __restrict__ W1, const float* __restrict__ b1, float* __restrict__ C1,
    const float* __restrict__ W2, const float* __restrict__ b2, float* __restrict__ C2)
{
    const float* W;
    const float* bias;
    float*       C;
    if (blockIdx.z == 0)      { W = W0; bias = b0; C = C0; }
    else if (blockIdx.z == 1) { W = W1; bias = b1; C = C1; }
    else                      { W = W2; bias = b2; C = C2; }

    __shared__ _Float16 As[128 * 40];   // A tile, row-major [m][k]   (10.0 KB)
    __shared__ _Float16 Bs[64 * 40];    // W tile transposed [n][k]   ( 5.0 KB)

    const int t    = threadIdx.x;
    const int lane = t & 31;
    const int wave = t >> 5;       // 0..7  -> M strip
    const int lh   = lane & 15;    // fragment row/col selector
    const int lg   = lane >> 4;    // fragment K-group selector

    const int m0 = blockIdx.y * 128;
    const int n0 = blockIdx.x * 64;

    // cooperative-load coordinates
    const int arow = t >> 1;          // 0..127
    const int acol = (t & 1) * 16;    // 0 or 16
    const int krow = t >> 3;          // 0..31 (k)
    const int ncol = (t & 7) * 8;     // 0..56 (n)

    v8f acc0 = {}, acc1 = {}, acc2 = {}, acc3 = {};

    for (int k0 = 0; k0 < DMODEL; k0 += 32) {
        // ---- A tile: 16 consecutive fp32 -> 16 halves, two 16B LDS stores
        const float* ap = A + (size_t)(m0 + arow) * DMODEL + k0 + acol;
        float4 x0 = *(const float4*)(ap);
        float4 x1 = *(const float4*)(ap + 4);
        float4 x2 = *(const float4*)(ap + 8);
        float4 x3 = *(const float4*)(ap + 12);
        _Float16* as = &As[arow * 40 + acol];
        as[0]  = (_Float16)x0.x; as[1]  = (_Float16)x0.y;
        as[2]  = (_Float16)x0.z; as[3]  = (_Float16)x0.w;
        as[4]  = (_Float16)x1.x; as[5]  = (_Float16)x1.y;
        as[6]  = (_Float16)x1.z; as[7]  = (_Float16)x1.w;
        as[8]  = (_Float16)x2.x; as[9]  = (_Float16)x2.y;
        as[10] = (_Float16)x2.z; as[11] = (_Float16)x2.w;
        as[12] = (_Float16)x3.x; as[13] = (_Float16)x3.y;
        as[14] = (_Float16)x3.z; as[15] = (_Float16)x3.w;

        // ---- W tile: coalesced read along n, transposed store [n][k]
        const float* wp = W + (size_t)(k0 + krow) * DMODEL + n0 + ncol;
        float4 w0 = *(const float4*)(wp);
        float4 w1 = *(const float4*)(wp + 4);
        Bs[(ncol + 0) * 40 + krow] = (_Float16)w0.x;
        Bs[(ncol + 1) * 40 + krow] = (_Float16)w0.y;
        Bs[(ncol + 2) * 40 + krow] = (_Float16)w0.z;
        Bs[(ncol + 3) * 40 + krow] = (_Float16)w0.w;
        Bs[(ncol + 4) * 40 + krow] = (_Float16)w1.x;
        Bs[(ncol + 5) * 40 + krow] = (_Float16)w1.y;
        Bs[(ncol + 6) * 40 + krow] = (_Float16)w1.z;
        Bs[(ncol + 7) * 40 + krow] = (_Float16)w1.w;

        if (k0 + 32 < DMODEL) {
            __builtin_prefetch(ap + 32, 0, 3);                 // global_prefetch_b8
            __builtin_prefetch(wp + 32 * DMODEL, 0, 3);
        }
        __syncthreads();

        // ---- fragments per CDNA5 16-bit WMMA VGPR layout
        // A: lane l -> row m0+wave*16+lh ; halves i: K = 8*lg + 16*(i>=8) + (i&7)
        const int aoff = (wave * 16 + lh) * 40 + 8 * lg;
        union { uint4 u[2]; v16h v; } fa;
        fa.u[0] = *(const uint4*)&As[aoff];
        fa.u[1] = *(const uint4*)&As[aoff + 16];

        // B: lane l -> col nt*16+lh ; halves i: K = 8*lg + 16*(i>=8) + (i&7)
        union { uint4 u[2]; v16h v; } fb;
#define B_FRAG(nt)                                             \
        {                                                      \
            const int boff = ((nt) * 16 + lh) * 40 + 8 * lg;   \
            fb.u[0] = *(const uint4*)&Bs[boff];                \
            fb.u[1] = *(const uint4*)&Bs[boff + 16];           \
        }
        B_FRAG(0);
        acc0 = __builtin_amdgcn_wmma_f32_16x16x32_f16(false, fa.v, false, fb.v,
                                                      (short)0, acc0, false, false);
        B_FRAG(1);
        acc1 = __builtin_amdgcn_wmma_f32_16x16x32_f16(false, fa.v, false, fb.v,
                                                      (short)0, acc1, false, false);
        B_FRAG(2);
        acc2 = __builtin_amdgcn_wmma_f32_16x16x32_f16(false, fa.v, false, fb.v,
                                                      (short)0, acc2, false, false);
        B_FRAG(3);
        acc3 = __builtin_amdgcn_wmma_f32_16x16x32_f16(false, fa.v, false, fb.v,
                                                      (short)0, acc3, false, false);
#undef B_FRAG
        __syncthreads();
    }

    // C/D layout: VGPR r -> M = r + 8*lg, N = lh
    const int rbase = m0 + wave * 16 + 8 * lg;
#define STORE_TILE(nt, accv)                                              \
    {                                                                     \
        const int col  = n0 + (nt) * 16 + lh;                             \
        const float bn = bias[col];                                       \
        _Pragma("unroll")                                                 \
        for (int r = 0; r < 8; ++r)                                       \
            C[(size_t)(rbase + r) * DMODEL + col] = (accv)[r] + bn;       \
    }
    STORE_TILE(0, acc0);
    STORE_TILE(1, acc1);
    STORE_TILE(2, acc2);
    STORE_TILE(3, acc3);
#undef STORE_TILE
}

// ---------------------------------------------------------------------------
// Vsum[b,h,d] = sum_j V[b,j,h*64+d]  (constant-background part of the
// softmax-weighted sum). 64 blocks x 64 threads, coalesced column sums.
// ---------------------------------------------------------------------------
__global__ __launch_bounds__(64) void vsum_kernel(const float* __restrict__ V,
                                                  float* __restrict__ Vsum)
{
    const int bh = blockIdx.x;            // 0..63
    const int b  = bh >> 4;
    const int h  = bh & 15;
    const int d  = threadIdx.x;           // 0..63
    const float* vp = V + (size_t)b * SEQ * DMODEL + h * DEPTH + d;
    float acc = 0.f;
    for (int j = 0; j < SEQ; ++j) acc += vp[(size_t)j * DMODEL];
    Vsum[bh * DEPTH + d] = acc;
}

// ---------------------------------------------------------------------------
// One wave per attention row (b,h,i): 32 neighbor dots, softmax folding the
// 992 implicit zero-score entries analytically, dense 4KB attn row built in
// LDS then drained with async LDS->global stores (ASYNCcnt path), and the
// context vector via ctx_i = base*Vsum + sum_w (p_w - base) * v_{g_w}.
// Block = 128 threads = 4 waves = 4 rows.
// ---------------------------------------------------------------------------
__global__ __launch_bounds__(128) void sparse_attn_kernel(
    const float* __restrict__ Q, const float* __restrict__ K,
    const float* __restrict__ V, const float* __restrict__ Vsum,
    const int* __restrict__ graph,
    float* __restrict__ attn, float* __restrict__ ctx)
{
    __shared__ float rows[4 * SEQ];

    const int lane = threadIdx.x & 31;
    const int wave = threadIdx.x >> 5;
    const int rid  = blockIdx.x * 4 + wave;      // (b*16+h)*1024 + i
    const int b    = rid >> 14;
    const int h    = (rid >> 10) & 15;
    const int i    = rid & (SEQ - 1);

    const int g = graph[i * GWIDTH + lane];

    // dot(q_i, k_g) over depth=64, fp32
    const float* qp = Q + ((size_t)b * SEQ + i) * DMODEL + h * DEPTH;
    const float* kp = K + ((size_t)b * SEQ + g) * DMODEL + h * DEPTH;
    float dot = 0.f;
#pragma unroll
    for (int t2 = 0; t2 < DEPTH / 4; ++t2) {
        float4 qv = ((const float4*)qp)[t2];
        float4 kv = ((const float4*)kp)[t2];
        dot += qv.x * kv.x + qv.y * kv.y + qv.z * kv.z + qv.w * kv.w;
    }
    dot *= 0.125f;                                  // 1/sqrt(64)

    // softmax over {32 dots} U {992 zeros}
    float m = dot;
#pragma unroll
    for (int off = 16; off; off >>= 1) m = fmaxf(m, __shfl_xor(m, off, 32));
    m = fmaxf(m, 0.f);                              // zeros participate in max
    const float e = __expf(dot - m);
    float s = e;
#pragma unroll
    for (int off = 16; off; off >>= 1) s += __shfl_xor(s, off, 32);
    const float zb   = __expf(-m);                  // each zero-score term
    const float Z    = s + (float)(SEQ - GWIDTH) * zb;
    const float inv  = 1.f / Z;
    const float base = zb * inv;                    // non-graph attn value
    const float p    = e  * inv;                    // this lane's graph value

    // dense attn row: fill base, scatter probs (same-wave DS ops are in-order)
    float* row = &rows[wave * SEQ];
    const float4 bv = make_float4(base, base, base, base);
#pragma unroll
    for (int j = lane * 4; j < SEQ; j += 128) *(float4*)&row[j] = bv;
    row[g] = p;

    float* arow = attn + (size_t)rid * SEQ;
#if USE_ASYNC_ATTN
    // DScnt and ASYNCcnt are unordered: make the DS stores land before the
    // async engine reads LDS (barrier implies s_wait_dscnt 0).
    __syncthreads();
#pragma unroll
    for (int j = lane * 4; j < SEQ; j += 128)
        __builtin_amdgcn_global_store_async_from_lds_b128(
            (__attribute__((address_space(1))) v4i*)(arow + j),
            (__attribute__((address_space(3))) v4i*)(row + j),
            0, 0);
#else
#pragma unroll
    for (int j = lane * 4; j < SEQ; j += 128)
        *(float4*)&arow[j] = *(const float4*)&row[j];
#endif

    // context: lane owns dims d=lane and d=lane+32 (overlaps async drain)
    const float* vs = Vsum + (b * NHEADS + h) * DEPTH;
    float acc0 = base * vs[lane];
    float acc1 = base * vs[lane + 32];
    const float coef = p - base;
#pragma unroll 4
    for (int w = 0; w < GWIDTH; ++w) {
        const float cw = __shfl(coef, w, 32);
        const int   gw = __shfl(g, w, 32);
        const float* vp = V + ((size_t)b * SEQ + gw) * DMODEL + h * DEPTH;
        acc0 += cw * vp[lane];
        acc1 += cw * vp[lane + 32];
    }
    float* cp = ctx + ((size_t)b * SEQ + i) * DMODEL + h * DEPTH;
    cp[lane]      = acc0;
    cp[lane + 32] = acc1;

#if USE_ASYNC_ATTN
    // LDS row must stay live until the async drain completes.
    __builtin_amdgcn_s_wait_asynccnt(0);
#endif
}

// ---------------------------------------------------------------------------
extern "C" void kernel_launch(void* const* d_in, const int* in_sizes, int n_in,
                              void* d_out, int out_size, void* d_ws, size_t ws_size,
                              hipStream_t stream)
{
    const float* hidden = (const float*)d_in[0];
    const float* wq = (const float*)d_in[1]; const float* bq = (const float*)d_in[2];
    const float* wk = (const float*)d_in[3]; const float* bk = (const float*)d_in[4];
    const float* wv = (const float*)d_in[5]; const float* bv = (const float*)d_in[6];
    const float* wo = (const float*)d_in[7]; const float* bo = (const float*)d_in[8];
    const int*   graph = (const int*)d_in[9];

    float* out  = (float*)d_out;                                   // (B,N,D)
    float* attn = out + (size_t)BATCH * SEQ * DMODEL;              // (B,H,N,N)

    const size_t MT = (size_t)BATCH * SEQ;        // 4096 rows
    float* ws   = (float*)d_ws;
    float* Q    = ws;
    float* Km   = ws + 1 * MT * DMODEL;
    float* Vm   = ws + 2 * MT * DMODEL;
    float* ctx  = ws + 3 * MT * DMODEL;
    float* Vsum = ws + 4 * MT * DMODEL;           // + 4096 floats

    // Q,K,V projections fused into one launch (grid.z selects weight).
    dim3 gqkv(DMODEL / 64, MT / 128, 3);
    gemm_f16_wmma<<<gqkv, 256, 0, stream>>>(hidden,
                                            wq, bq, Q,
                                            wk, bk, Km,
                                            wv, bv, Vm);

    vsum_kernel<<<BATCH * NHEADS, DEPTH, 0, stream>>>(Vm, Vsum);

    // 4*16*1024 rows, 4 rows (waves) per block
    sparse_attn_kernel<<<(BATCH * NHEADS * SEQ) / 4, 128, 0, stream>>>(
        Q, Km, Vm, Vsum, graph, attn, ctx);

    // output projection
    dim3 go(DMODEL / 64, MT / 128, 1);
    gemm_f16_wmma<<<go, 256, 0, stream>>>(ctx,
                                          wo, bo, out,
                                          wo, bo, out,
                                          wo, bo, out);
}
